// TransformerBlock_86260123173353
// MI455X (gfx1250) — matrix operand order repaired and checked by reading, not compiled, not run on hardware
//
#include <hip/hip_runtime.h>
#include <math.h>

#define D_MODEL 1024
#define N_HEADS 16
#define HEAD_DIM 64
#define D_INNER 4096
#define BATCH 2
#define SEQ 2048
#define TOKENS (BATCH * SEQ)

typedef __bf16 bf16_t;
typedef bf16_t v16bf __attribute__((ext_vector_type(16)));
typedef float  v8f   __attribute__((ext_vector_type(8)));
typedef unsigned short ushort8  __attribute__((ext_vector_type(8)));
typedef float  float4v __attribute__((ext_vector_type(4)));

union BFrag16 { ushort8 u[2]; v16bf v; unsigned short s[16]; };

__device__ __forceinline__ unsigned short f32_to_bf16(float f) {
    union { float f; unsigned u; } v; v.f = f;
    unsigned r = v.u + 0x7FFFu + ((v.u >> 16) & 1u);   // round-to-nearest-even
    return (unsigned short)(r >> 16);
}

__device__ __forceinline__ v8f v8f_zero() {
    v8f z;
#pragma unroll
    for (int j = 0; j < 8; ++j) z[j] = 0.f;
    return z;
}

// Async 16B copy global -> LDS (ASYNCcnt-tracked DMA, no VGPR data path).
__device__ __forceinline__ void async_copy16(unsigned lds_off, const void* gptr) {
    asm volatile("global_load_async_to_lds_b128 %0, %1, off"
                 :: "v"(lds_off), "v"((unsigned long long)(size_t)gptr)
                 : "memory");
}

// ---------------------------------------------------------------------------
// Weight convert+transpose: W[K][N] f32 -> Wt[N][K] bf16 (one-time, cheap)
// ---------------------------------------------------------------------------
__global__ __launch_bounds__(256) void transpose_to_bf16(
    const float* __restrict__ W, unsigned short* __restrict__ Wt, int K, int N) {
    __shared__ unsigned short tile[32][33];
    const int bx = blockIdx.x * 32;  // N
    const int by = blockIdx.y * 32;  // K
    const int tx = threadIdx.x & 31;
    const int ty = threadIdx.x >> 5;  // 0..7
#pragma unroll
    for (int i = ty; i < 32; i += 8)
        tile[i][tx] = f32_to_bf16(W[(size_t)(by + i) * N + bx + tx]);
    __syncthreads();
#pragma unroll
    for (int i = ty; i < 32; i += 8)
        Wt[(size_t)(bx + i) * K + by + tx] = tile[tx][i];
}

// ---------------------------------------------------------------------------
// RMSNorm: one block per token; f32 in, bf16 out
// ---------------------------------------------------------------------------
__global__ __launch_bounds__(256) void rmsnorm_kernel(const float* __restrict__ X,
                                                      unsigned short* __restrict__ Y) {
    __shared__ float red[8];
    const int t = blockIdx.x;
    const float* x = X + (size_t)t * D_MODEL;
    unsigned short* y = Y + (size_t)t * D_MODEL;
    const int tid = threadIdx.x;
    float s = 0.f;
    for (int j = tid; j < D_MODEL; j += 256) { float v = x[j]; s += v * v; }
#pragma unroll
    for (int m = 16; m >= 1; m >>= 1) s += __shfl_xor(s, m);
    const int lane = tid & 31, wave = tid >> 5;
    if (lane == 0) red[wave] = s;
    __syncthreads();
    if (wave == 0) {
        float v = (lane < 8) ? red[lane] : 0.f;
#pragma unroll
        for (int m = 4; m >= 1; m >>= 1) v += __shfl_xor(v, m);
        if (lane == 0) red[0] = v;
    }
    __syncthreads();
    const float inv = rsqrtf(red[0] / (float)D_MODEL + 1e-6f);
    for (int j = tid; j < D_MODEL; j += 256) y[j] = f32_to_bf16(x[j] * inv);
}

// ---------------------------------------------------------------------------
// bf16 WMMA GEMM: C[M,N] = A[M,K] * Bt[N,K]^T. A,Bt bf16 row-major.
// 128x128 block tile, BK=32, double-buffered LDS filled with async DMA.
// 256 threads = 8 waves (2x4); each wave computes 64x32 (eight 16x16 tiles).
// MODE 1: bf16 out into [B,H,T,64] head layout
// MODE 2: f32 out = acc + Res
// MODE 3: bf16 out = gelu(acc)
// ---------------------------------------------------------------------------
#define LDSS 40                 // padded LDS row stride in halves (80B)
#define TILE_HALVES (128 * LDSS)

template <int MODE>
__global__ __launch_bounds__(256) void gemm_wmma_bf16(
    const unsigned short* __restrict__ A, const unsigned short* __restrict__ Bt,
    void* __restrict__ Out, const float* __restrict__ Res,
    int M, int N, int K) {
    __shared__ unsigned short As[2][TILE_HALVES];
    __shared__ unsigned short Bs[2][TILE_HALVES];

    const int tid  = threadIdx.x;
    const int lane = tid & 31;
    const int wave = tid >> 5;
    const int half = lane >> 4;
    const int r    = lane & 15;
    const int wm   = wave >> 2;  // 0..1
    const int wn   = wave & 3;   // 0..3
    const int m0   = blockIdx.y * 128;
    const int n0   = blockIdx.x * 128;

    v8f acc[4][2];
#pragma unroll
    for (int mi = 0; mi < 4; ++mi)
#pragma unroll
        for (int ni = 0; ni < 2; ++ni) acc[mi][ni] = v8f_zero();

    // chunk mapping: 512 x 16B chunks per tile, 2 per thread per tile
    const int row0 = tid >> 2;           // 0..63
    const int seg  = (tid & 3) * 8;      // halves 0,8,16,24

    auto stage = [&](int buf, int k0) {
        const unsigned as_off = (unsigned)(size_t)(&As[buf][0]);
        const unsigned bs_off = (unsigned)(size_t)(&Bs[buf][0]);
#pragma unroll
        for (int it = 0; it < 2; ++it) {
            const int row = row0 + it * 64;
            const unsigned loff = (unsigned)(row * LDSS + seg) * 2u;
            async_copy16(as_off + loff, A  + (size_t)(m0 + row) * K + k0 + seg);
            async_copy16(bs_off + loff, Bt + (size_t)(n0 + row) * K + k0 + seg);
        }
    };

    const int nsteps = K >> 5;
    stage(0, 0);
    for (int s = 0; s < nsteps; ++s) {
        const int buf = s & 1;
        if (s + 1 < nsteps) {
            stage(buf ^ 1, (s + 1) << 5);
            asm volatile("s_wait_asynccnt 0x4" ::: "memory");  // stage s landed
        } else {
            asm volatile("s_wait_asynccnt 0x0" ::: "memory");
        }
        __syncthreads();

        BFrag16 af[4], bf[2];
#pragma unroll
        for (int mi = 0; mi < 4; ++mi) {
            const unsigned short* p = &As[buf][(wm * 64 + mi * 16 + r) * LDSS + half * 8];
            af[mi].u[0] = *(const ushort8*)(p);
            af[mi].u[1] = *(const ushort8*)(p + 16);
        }
#pragma unroll
        for (int ni = 0; ni < 2; ++ni) {
            const unsigned short* p = &Bs[buf][(wn * 32 + ni * 16 + r) * LDSS + half * 8];
            bf[ni].u[0] = *(const ushort8*)(p);
            bf[ni].u[1] = *(const ushort8*)(p + 16);
        }
#pragma unroll
        for (int mi = 0; mi < 4; ++mi)
#pragma unroll
            for (int ni = 0; ni < 2; ++ni)
                acc[mi][ni] = __builtin_amdgcn_wmma_f32_16x16x32_bf16(
                    false, af[mi].v, false, bf[ni].v, (short)0, acc[mi][ni],
                    false, false);
        __syncthreads();
    }

#pragma unroll
    for (int mi = 0; mi < 4; ++mi) {
#pragma unroll
        for (int ni = 0; ni < 2; ++ni) {
#pragma unroll
            for (int i = 0; i < 8; ++i) {
                const int m = m0 + wm * 64 + mi * 16 + i + half * 8;
                const int n = n0 + wn * 32 + ni * 16 + r;
                const float v = acc[mi][ni][i];
                if (MODE == 1) {
                    const int b = m / SEQ, t = m % SEQ;
                    const int h = n / HEAD_DIM, d = n % HEAD_DIM;
                    ((unsigned short*)Out)[(((size_t)(b * N_HEADS + h)) * SEQ + t) * HEAD_DIM + d] =
                        f32_to_bf16(v);
                } else if (MODE == 2) {
                    ((float*)Out)[(size_t)m * N + n] = v + Res[(size_t)m * N + n];
                } else {
                    const float x = v;
                    const float g = 0.5f * x *
                        (1.0f + tanhf(0.7978845608028654f * (x + 0.044715f * x * x * x)));
                    ((unsigned short*)Out)[(size_t)m * N + n] = f32_to_bf16(g);
                }
            }
        }
    }
}

// ---------------------------------------------------------------------------
// Flash attention: Q,K,V bf16 [B*H, T, 64] -> Out bf16 [B, T, D_MODEL].
// 4 waves x 16 query rows per block; kv in chunks of 32 (= bf16 WMMA K).
// ---------------------------------------------------------------------------
__global__ __launch_bounds__(128) void flash_attn_wmma(
    const unsigned short* __restrict__ Q, const unsigned short* __restrict__ Km,
    const unsigned short* __restrict__ Vm, unsigned short* __restrict__ Out) {
    __shared__ unsigned short Vt[64 * LDSS];      // V^T tile: [d][key]
    __shared__ unsigned short Ps[4 * 16 * LDSS];  // per-wave P scratch

    const int tid  = threadIdx.x;
    const int lane = tid & 31;
    const int wave = tid >> 5;
    const int half = lane >> 4;
    const int r    = lane & 15;
    const int q0   = blockIdx.x * 64;
    const int bh   = blockIdx.y;
    const size_t base = (size_t)bh * SEQ * HEAD_DIM;
    const int b = bh / N_HEADS;
    const int h = bh % N_HEADS;
    const int qw = q0 + wave * 16;

    BFrag16 qa[2];
#pragma unroll
    for (int ks = 0; ks < 2; ++ks) {
        const unsigned short* p = Q + base + (size_t)(qw + r) * HEAD_DIM + ks * 32 + half * 8;
        qa[ks].u[0] = *(const ushort8*)(p);
        qa[ks].u[1] = *(const ushort8*)(p + 16);
    }

    v8f o[4];
#pragma unroll
    for (int nt = 0; nt < 4; ++nt) o[nt] = v8f_zero();
    float mv[8], lv[8];
#pragma unroll
    for (int i = 0; i < 8; ++i) { mv[i] = -1e30f; lv[i] = 0.f; }
    const float scale = 0.125f;

    for (int kv0 = 0; kv0 < q0 + 64; kv0 += 32) {
        {   // cooperative stage V^T
            const int key = tid >> 2;
            const int dg  = (tid & 3) * 16;
            const unsigned short* p = Vm + base + (size_t)(kv0 + key) * HEAD_DIM + dg;
            BFrag16 tmp;
            tmp.u[0] = *(const ushort8*)(p);
            tmp.u[1] = *(const ushort8*)(p + 8);
#pragma unroll
            for (int j = 0; j < 16; ++j) Vt[(dg + j) * LDSS + key] = tmp.s[j];
        }
        __syncthreads();

        v8f s[2];
#pragma unroll
        for (int ni = 0; ni < 2; ++ni) {
            s[ni] = v8f_zero();
#pragma unroll
            for (int ks = 0; ks < 2; ++ks) {
                BFrag16 kf;
                const unsigned short* p =
                    Km + base + (size_t)(kv0 + ni * 16 + r) * HEAD_DIM + ks * 32 + half * 8;
                kf.u[0] = *(const ushort8*)(p);
                kf.u[1] = *(const ushort8*)(p + 16);
                s[ni] = __builtin_amdgcn_wmma_f32_16x16x32_bf16(
                    false, qa[ks].v, false, kf.v, (short)0, s[ni], false, false);
            }
        }

        float p0[8], p1[8];
#pragma unroll
        for (int i = 0; i < 8; ++i) {
            const int qidx = qw + i + half * 8;
            float s0 = s[0][i] * scale + ((kv0 + r > qidx) ? -1e30f : 0.f);
            float s1 = s[1][i] * scale + ((kv0 + 16 + r > qidx) ? -1e30f : 0.f);
            float rm = fmaxf(s0, s1);
            rm = fmaxf(rm, __shfl_xor(rm, 1));
            rm = fmaxf(rm, __shfl_xor(rm, 2));
            rm = fmaxf(rm, __shfl_xor(rm, 4));
            rm = fmaxf(rm, __shfl_xor(rm, 8));
            const float mn    = fmaxf(mv[i], rm);
            const float alpha = __expf(mv[i] - mn);
            const float e0 = __expf(s0 - mn);
            const float e1 = __expf(s1 - mn);
            float rs = e0 + e1;
            rs += __shfl_xor(rs, 1);
            rs += __shfl_xor(rs, 2);
            rs += __shfl_xor(rs, 4);
            rs += __shfl_xor(rs, 8);
            lv[i] = lv[i] * alpha + rs;
            mv[i] = mn;
            p0[i] = e0; p1[i] = e1;
#pragma unroll
            for (int nt = 0; nt < 4; ++nt) o[nt][i] *= alpha;
        }

        unsigned short* ps = &Ps[wave * 16 * LDSS];
#pragma unroll
        for (int i = 0; i < 8; ++i) {
            ps[(i + half * 8) * LDSS + r]      = f32_to_bf16(p0[i]);
            ps[(i + half * 8) * LDSS + 16 + r] = f32_to_bf16(p1[i]);
        }
        asm volatile("s_wait_dscnt 0x0" ::: "memory");
        BFrag16 pa;
        {
            const unsigned short* p = ps + r * LDSS + half * 8;
            pa.u[0] = *(const ushort8*)(p);
            pa.u[1] = *(const ushort8*)(p + 16);
        }
#pragma unroll
        for (int nt = 0; nt < 4; ++nt) {
            BFrag16 vb;
            const unsigned short* p = &Vt[(nt * 16 + r) * LDSS + half * 8];
            vb.u[0] = *(const ushort8*)(p);
            vb.u[1] = *(const ushort8*)(p + 16);
            o[nt] = __builtin_amdgcn_wmma_f32_16x16x32_bf16(
                false, pa.v, false, vb.v, (short)0, o[nt], false, false);
        }
        __syncthreads();
    }

#pragma unroll
    for (int i = 0; i < 8; ++i) {
        const float inv = 1.0f / lv[i];
        const int t = qw + i + half * 8;
#pragma unroll
        for (int nt = 0; nt < 4; ++nt) {
            const int col = h * HEAD_DIM + nt * 16 + r;
            Out[((size_t)(b * SEQ + t)) * D_MODEL + col] = f32_to_bf16(o[nt][i] * inv);
        }
    }
}

// ---------------------------------------------------------------------------
// Host launcher
// ---------------------------------------------------------------------------
extern "C" void kernel_launch(void* const* d_in, const int* in_sizes, int n_in,
                              void* d_out, int out_size, void* d_ws, size_t ws_size,
                              hipStream_t stream) {
    const float* x  = (const float*)d_in[0];
    const float* Wq = (const float*)d_in[2];
    const float* Wk = (const float*)d_in[3];
    const float* Wv = (const float*)d_in[4];
    const float* Wo = (const float*)d_in[5];
    const float* W1 = (const float*)d_in[6];
    const float* W2 = (const float*)d_in[7];
    float* out = (float*)d_out;

    typedef unsigned short u16;
    char* ws = (char*)d_ws;
    size_t off = 0;
    u16* wqt = (u16*)(ws + off); off += (size_t)D_MODEL * D_MODEL * 2;
    u16* wkt = (u16*)(ws + off); off += (size_t)D_MODEL * D_MODEL * 2;
    u16* wvt = (u16*)(ws + off); off += (size_t)D_MODEL * D_MODEL * 2;
    u16* wot = (u16*)(ws + off); off += (size_t)D_MODEL * D_MODEL * 2;
    u16* w1t = (u16*)(ws + off); off += (size_t)D_MODEL * D_INNER * 2;
    u16* w2t = (u16*)(ws + off); off += (size_t)D_MODEL * D_INNER * 2;
    u16* h   = (u16*)(ws + off); off += (size_t)TOKENS * D_MODEL * 2;
    u16* qb  = (u16*)(ws + off); off += (size_t)TOKENS * D_MODEL * 2;
    u16* kb  = (u16*)(ws + off); off += (size_t)TOKENS * D_MODEL * 2;
    u16* vbuf= (u16*)(ws + off); off += (size_t)TOKENS * D_MODEL * 2;
    u16* attn= (u16*)(ws + off); off += (size_t)TOKENS * D_MODEL * 2;
    float* x1= (float*)(ws + off); off += (size_t)TOKENS * D_MODEL * 4;
    u16* h2  = (u16*)(ws + off); off += (size_t)TOKENS * D_MODEL * 2;
    u16* f1  = (u16*)(ws + off); off += (size_t)TOKENS * D_INNER * 2;
    (void)ws_size; (void)in_sizes; (void)n_in; (void)out_size;

    // 0. weights -> bf16, transposed to [N][K]
    dim3 gt(D_MODEL / 32, D_MODEL / 32);
    transpose_to_bf16<<<gt, 256, 0, stream>>>(Wq, wqt, D_MODEL, D_MODEL);
    transpose_to_bf16<<<gt, 256, 0, stream>>>(Wk, wkt, D_MODEL, D_MODEL);
    transpose_to_bf16<<<gt, 256, 0, stream>>>(Wv, wvt, D_MODEL, D_MODEL);
    transpose_to_bf16<<<gt, 256, 0, stream>>>(Wo, wot, D_MODEL, D_MODEL);
    dim3 gt1(D_INNER / 32, D_MODEL / 32);
    transpose_to_bf16<<<gt1, 256, 0, stream>>>(W1, w1t, D_MODEL, D_INNER);
    dim3 gt2(D_MODEL / 32, D_INNER / 32);
    transpose_to_bf16<<<gt2, 256, 0, stream>>>(W2, w2t, D_INNER, D_MODEL);

    // 1. h = rmsnorm(x) -> bf16
    rmsnorm_kernel<<<TOKENS, 256, 0, stream>>>(x, h);

    // 2. Q,K,V = h @ W{q,k,v} -> bf16 [B,H,T,64]
    dim3 g1(D_MODEL / 128, TOKENS / 128);
    gemm_wmma_bf16<1><<<g1, 256, 0, stream>>>(h, wqt, qb, nullptr, TOKENS, D_MODEL, D_MODEL);
    gemm_wmma_bf16<1><<<g1, 256, 0, stream>>>(h, wkt, kb, nullptr, TOKENS, D_MODEL, D_MODEL);
    gemm_wmma_bf16<1><<<g1, 256, 0, stream>>>(h, wvt, vbuf, nullptr, TOKENS, D_MODEL, D_MODEL);

    // 3. attn = softmax(QK^T * scale, causal) V -> bf16 [B,T,D]
    dim3 ga(SEQ / 64, BATCH * N_HEADS);
    flash_attn_wmma<<<ga, 128, 0, stream>>>(qb, kb, vbuf, attn);

    // 4. x1 = x + attn @ Wo  (f32)
    gemm_wmma_bf16<2><<<g1, 256, 0, stream>>>(attn, wot, x1, x, TOKENS, D_MODEL, D_MODEL);

    // 5. h2 = rmsnorm(x1) -> bf16
    rmsnorm_kernel<<<TOKENS, 256, 0, stream>>>(x1, h2);

    // 6. f1 = gelu(h2 @ W1) -> bf16
    dim3 g2(D_INNER / 128, TOKENS / 128);
    gemm_wmma_bf16<3><<<g2, 256, 0, stream>>>(h2, w1t, f1, nullptr, TOKENS, D_INNER, D_MODEL);

    // 7. out = x1 + f1 @ W2  (f32)
    gemm_wmma_bf16<2><<<g1, 256, 0, stream>>>(f1, w2t, out, x1, TOKENS, D_MODEL, D_INNER);
}
